// RobustFill_44925357916632
// MI455X (gfx1250) — matrix operand order repaired and checked by reading, not compiled, not run on hardware
//
#include <hip/hip_runtime.h>
#include <hip/hip_bf16.h>

// Problem dims (fixed by reference)
#define TDIM  64
#define SDIM  128
#define BDIM  256
#define HDIM  512
#define INDIM 128
#define G4H   2048   // 4*H
#define KIN   640    // IN + H

typedef __attribute__((ext_vector_type(16))) __bf16 v16bf;
typedef __attribute__((ext_vector_type(8)))  __bf16 v8bf;
typedef __attribute__((ext_vector_type(8)))  float  v8f;

__device__ __forceinline__ __bf16 f2bf(float f) {
  union { float f; unsigned u; } in; in.f = f;
  unsigned u = in.u;
  unsigned r = u + 0x7FFFu + ((u >> 16) & 1u);   // round-to-nearest-even
  unsigned short s = (unsigned short)(r >> 16);
  union { unsigned short s; __bf16 b; } out; out.s = s;
  return out.b;
}

__device__ __forceinline__ float sigmoidf_fast(float x) {
  return 1.0f / (1.0f + __expf(-x));
}

// ---------------------------------------------------------------------------
// One-time prep kernels
// ---------------------------------------------------------------------------
__global__ void cast_bf16_kernel(const float* __restrict__ src,
                                 __bf16* __restrict__ dst, int n) {
  int i = blockIdx.x * blockDim.x + threadIdx.x;
  if (i < n) dst[i] = f2bf(src[i]);
}

__global__ void bias_init_kernel(const float* __restrict__ bi,
                                 const float* __restrict__ bh,
                                 float* __restrict__ bc, int n) {
  int i = blockIdx.x * blockDim.x + threadIdx.x;
  if (i < n) bc[i] = bi[i] + bh[i];
}

__global__ void state_init_kernel(const float* __restrict__ h0,
                                  const float* __restrict__ c0,
                                  float* __restrict__ h, float* __restrict__ c,
                                  __bf16* __restrict__ hbf, int n) {
  int i = blockIdx.x * blockDim.x + threadIdx.x;
  if (i < n) {
    float hv = h0[i];
    h[i] = hv;
    c[i] = c0[i];
    hbf[i] = f2bf(hv);
  }
}

// ---------------------------------------------------------------------------
// WMMA GEMM:  C[M=256, N] = A1(bf16, M x K1) * W1(bf16, N x K1)^T
//                         [+ A2(M x K2) * W2(N x K2)^T] + bias[N]
// Block = 256 threads = 8 waves; each wave owns one 16x16 output tile.
// Grid: (N/128, M/16). Fragments loaded straight from L2 in the ISA's
// per-lane layouts (16-bit A 16x32, B 32x16), no LDS staging needed.
// ---------------------------------------------------------------------------
__global__ __launch_bounds__(256)
void gemm_wmma_kernel(const __bf16* __restrict__ A1, const __bf16* __restrict__ W1, int K1,
                      const __bf16* __restrict__ A2, const __bf16* __restrict__ W2, int K2,
                      const float* __restrict__ bias, float* __restrict__ C, int N) {
  const int lane = threadIdx.x & 31;
  const int wave = threadIdx.x >> 5;
  const int tm   = blockIdx.y;                 // M tile (16 rows)
  const int tn   = blockIdx.x * 8 + wave;      // N tile (16 cols)
  const int l15  = lane & 15;
  const int mrow = tm * 16 + l15;              // A row this lane contributes
  const int wrow = tn * 16 + l15;              // W row (== B-matrix column)
  const int kA   = (lane >> 4) * 8;            // A-frag K sub-offset per ISA layout
  const int kB   = (lane >> 4) * 8;            // B-frag K sub-offset per ISA layout

  v8f acc = {};

  {
    const __bf16* pa = A1 + (size_t)mrow * K1 + kA;
    const __bf16* pb = W1 + (size_t)wrow * K1 + kB;
#pragma unroll 4
    for (int k = 0; k < K1; k += 32) {
      v8bf a0 = *(const v8bf*)(pa + k);        // K = kA+0..7   (block 0)
      v8bf a1 = *(const v8bf*)(pa + k + 16);   // K = kA+16..23 (block 1)
      v8bf b0 = *(const v8bf*)(pb + k);        // K = kB+0..7
      v8bf b1 = *(const v8bf*)(pb + k + 16);   // K = kB+16..23
      v16bf af = __builtin_shufflevector(a0, a1, 0,1,2,3,4,5,6,7,8,9,10,11,12,13,14,15);
      v16bf bfv = __builtin_shufflevector(b0, b1, 0,1,2,3,4,5,6,7,8,9,10,11,12,13,14,15);
      acc = __builtin_amdgcn_wmma_f32_16x16x32_bf16(false, af, false, bfv,
                                                    (short)0, acc, false, false);
    }
  }
  if (K2 > 0) {
    const __bf16* pa = A2 + (size_t)mrow * K2 + kA;
    const __bf16* pb = W2 + (size_t)wrow * K2 + kB;
#pragma unroll 4
    for (int k = 0; k < K2; k += 32) {
      v8bf a0 = *(const v8bf*)(pa + k);
      v8bf a1 = *(const v8bf*)(pa + k + 16);
      v8bf b0 = *(const v8bf*)(pb + k);
      v8bf b1 = *(const v8bf*)(pb + k + 16);
      v16bf af = __builtin_shufflevector(a0, a1, 0,1,2,3,4,5,6,7,8,9,10,11,12,13,14,15);
      v16bf bfv = __builtin_shufflevector(b0, b1, 0,1,2,3,4,5,6,7,8,9,10,11,12,13,14,15);
      acc = __builtin_amdgcn_wmma_f32_16x16x32_bf16(false, af, false, bfv,
                                                    (short)0, acc, false, false);
    }
  }

  const float bv = bias ? bias[tn * 16 + l15] : 0.0f;
  const int mbase = (lane >> 4) * 8;           // C/D layout: lanes 16-31 hold M=8..15
  float* pc = C + (size_t)(tm * 16 + mbase) * N + tn * 16 + l15;
#pragma unroll
  for (int i = 0; i < 8; ++i) pc[(size_t)i * N] = acc[i] + bv;
}

// ---------------------------------------------------------------------------
// Attention: per-batch block. scores = attended[s,b,:] . q[b,:], masked
// softmax over S, context = sum_s align[s] * attended[s,b,:]
// ---------------------------------------------------------------------------
__global__ __launch_bounds__(256)
void attention_kernel(const float* __restrict__ q,
                      const float* __restrict__ attended,
                      const int* __restrict__ seq_len,
                      float* __restrict__ ctx) {
  __shared__ float sq[HDIM];
  __shared__ float ss[SDIM];
  __shared__ float sa[SDIM];
  const int b   = blockIdx.x;
  const int tid = threadIdx.x;

  sq[tid]       = q[(size_t)b * HDIM + tid];
  sq[tid + 256] = q[(size_t)b * HDIM + tid + 256];
  __syncthreads();

  const int lane = tid & 31;
  const int wave = tid >> 5;
  for (int s = wave; s < SDIM; s += 8) {
    const float* att = attended + ((size_t)s * BDIM + b) * HDIM;
    float acc = 0.0f;
    for (int hh = lane; hh < HDIM; hh += 32) acc += att[hh] * sq[hh];
#pragma unroll
    for (int off = 16; off > 0; off >>= 1) acc += __shfl_down(acc, off);
    if (lane == 0) ss[s] = acc;
  }
  __syncthreads();

  const int L = seq_len[b];
  if (tid < SDIM && tid >= L) ss[tid] = -__builtin_inff();
  __syncthreads();

  // redundant per-thread scan (broadcast LDS reads, 128 elems — cheap)
  float mx = -__builtin_inff();
  for (int s = 0; s < SDIM; ++s) mx = fmaxf(mx, ss[s]);
  float sum = 0.0f;
  for (int s = 0; s < SDIM; ++s) sum += __expf(ss[s] - mx);
  const float inv = 1.0f / sum;
  if (tid < SDIM) sa[tid] = __expf(ss[tid] - mx) * inv;
  __syncthreads();

  float c0 = 0.0f, c1 = 0.0f;
  for (int s = 0; s < L; ++s) {
    const float w = sa[s];
    const float* att = attended + ((size_t)s * BDIM + b) * HDIM;
    c0 += w * att[tid];
    c1 += w * att[tid + 256];
  }
  ctx[(size_t)b * HDIM + tid]       = c0;
  ctx[(size_t)b * HDIM + tid + 256] = c1;
}

// ---------------------------------------------------------------------------
// inp_bf16[b, 0:128] = x_t[b,:] ; inp_bf16[b, 128:640] = context[b,:]
// ---------------------------------------------------------------------------
__global__ void build_inp_kernel(const float* __restrict__ x_t,
                                 const float* __restrict__ ctx,
                                 __bf16* __restrict__ inp) {
  int idx = blockIdx.x * blockDim.x + threadIdx.x;
  if (idx >= BDIM * KIN) return;
  int b = idx / KIN, j = idx - b * KIN;
  float v = (j < INDIM) ? x_t[b * INDIM + j] : ctx[(size_t)b * HDIM + (j - INDIM)];
  inp[idx] = f2bf(v);
}

// ---------------------------------------------------------------------------
// LSTM pointwise (torch gate order i,f,g,o); writes h, c, h_bf16, out[t]
// ---------------------------------------------------------------------------
__global__ __launch_bounds__(512)
void lstm_pointwise_kernel(const float* __restrict__ gates,
                           float* __restrict__ c, float* __restrict__ h,
                           __bf16* __restrict__ hbf, float* __restrict__ out_t) {
  int idx = blockIdx.x * 512 + threadIdx.x;   // BDIM*HDIM threads
  int b = idx >> 9, hh = idx & (HDIM - 1);
  const float* g = gates + (size_t)b * G4H;
  float gi = sigmoidf_fast(g[hh]);
  float gf = sigmoidf_fast(g[HDIM + hh]);
  float gg = tanhf(g[2 * HDIM + hh]);
  float go = sigmoidf_fast(g[3 * HDIM + hh]);
  float cn = gf * c[idx] + gi * gg;
  float hn = go * tanhf(cn);
  c[idx] = cn;
  h[idx] = hn;
  hbf[idx] = f2bf(hn);
  out_t[idx] = hn;
}

// ---------------------------------------------------------------------------
// Host launcher
// ---------------------------------------------------------------------------
extern "C" void kernel_launch(void* const* d_in, const int* in_sizes, int n_in,
                              void* d_out, int out_size, void* d_ws, size_t ws_size,
                              hipStream_t stream) {
  (void)in_sizes; (void)n_in; (void)out_size; (void)ws_size;
  const float* x        = (const float*)d_in[0];   // (T,B,IN)
  const float* h0       = (const float*)d_in[1];   // (B,H)
  const float* c0       = (const float*)d_in[2];   // (B,H)
  const float* attended = (const float*)d_in[3];   // (S,B,H)
  const int*   seqlen   = (const int*)d_in[4];     // (B,)
  const float* W_att    = (const float*)d_in[5];   // (H,H)
  const float* b_att    = (const float*)d_in[6];   // (H,)
  const float* W_ih     = (const float*)d_in[7];   // (4H, IN+H)
  const float* W_hh     = (const float*)d_in[8];   // (4H, H)
  const float* b_ih     = (const float*)d_in[9];   // (4H,)
  const float* b_hh     = (const float*)d_in[10];  // (4H,)
  float* out = (float*)d_out;                      // (T,B,H)

  // Workspace carve-up (256B aligned slices)
  char* ws = (char*)d_ws;
  size_t off = 0;
  auto carve = [&](size_t bytes) -> char* {
    char* p = ws + off;
    off = (off + bytes + 255) & ~(size_t)255;
    return p;
  };
  __bf16* wAttB = (__bf16*)carve((size_t)HDIM * HDIM * 2);
  __bf16* wIhB  = (__bf16*)carve((size_t)G4H * KIN * 2);
  __bf16* wHhB  = (__bf16*)carve((size_t)G4H * HDIM * 2);
  float*  biasC = (float*)carve((size_t)G4H * 4);
  float*  h     = (float*)carve((size_t)BDIM * HDIM * 4);
  float*  c     = (float*)carve((size_t)BDIM * HDIM * 4);
  __bf16* hB    = (__bf16*)carve((size_t)BDIM * HDIM * 2);
  float*  q     = (float*)carve((size_t)BDIM * HDIM * 4);
  float*  ctx   = (float*)carve((size_t)BDIM * HDIM * 4);
  __bf16* inpB  = (__bf16*)carve((size_t)BDIM * KIN * 2);
  float*  gates = (float*)carve((size_t)BDIM * G4H * 4);

  // ---- one-time prep (deterministic every call) ----
  {
    int n = HDIM * HDIM;
    cast_bf16_kernel<<<(n + 255) / 256, 256, 0, stream>>>(W_att, wAttB, n);
    n = G4H * KIN;
    cast_bf16_kernel<<<(n + 255) / 256, 256, 0, stream>>>(W_ih, wIhB, n);
    n = G4H * HDIM;
    cast_bf16_kernel<<<(n + 255) / 256, 256, 0, stream>>>(W_hh, wHhB, n);
    bias_init_kernel<<<(G4H + 255) / 256, 256, 0, stream>>>(b_ih, b_hh, biasC, G4H);
    n = BDIM * HDIM;
    state_init_kernel<<<(n + 255) / 256, 256, 0, stream>>>(h0, c0, h, c, hB, n);
  }

  // ---- sequential scan over T timesteps ----
  for (int t = 0; t < TDIM; ++t) {
    // q = h @ W_att^T + b_att        (WMMA, N=512)
    gemm_wmma_kernel<<<dim3(HDIM / 128, BDIM / 16), 256, 0, stream>>>(
        hB, wAttB, HDIM, nullptr, nullptr, 0, b_att, q, HDIM);

    // masked softmax attention -> context
    attention_kernel<<<BDIM, 256, 0, stream>>>(q, attended, seqlen, ctx);

    // inp = bf16(concat(x_t, context))
    build_inp_kernel<<<(BDIM * KIN + 255) / 256, 256, 0, stream>>>(
        x + (size_t)t * BDIM * INDIM, ctx, inpB);

    // gates = inp @ W_ih^T + h @ W_hh^T + (b_ih + b_hh)   (WMMA, N=2048, dual-K)
    gemm_wmma_kernel<<<dim3(G4H / 128, BDIM / 16), 256, 0, stream>>>(
        inpB, wIhB, KIN, hB, wHhB, HDIM, biasC, gates, G4H);

    // pointwise LSTM cell; updates h/c/h_bf16 and writes out[t]
    lstm_pointwise_kernel<<<BDIM, 512, 0, stream>>>(
        gates, c, h, hB, out + (size_t)t * BDIM * HDIM);
  }
}
